// DecoderWithAttention_justify_45268955300300
// MI455X (gfx1250) — matrix operand order repaired and checked by reading, not compiled, not run on hardware
//
#include <hip/hip_runtime.h>

// ---------------------------------------------------------------------------
// DecoderWithAttention for MI455X (gfx1250), wave32, bf16 WMMA w/ f32 accum.
// All GEMMs: C[MxN] = A[MxK](bf16) @ Bt[NxK](bf16)^T (+bias), fp32 out.
// ---------------------------------------------------------------------------

#define B_  64
#define P_  196
#define D_  512
#define T_  51
#define V_  10000

typedef __attribute__((ext_vector_type(16))) __bf16 v16bf;
typedef __attribute__((ext_vector_type(8)))  float  v8f;

__device__ __forceinline__ float sigf(float v) { return 1.f / (1.f + __expf(-v)); }

// ---------------------------------------------------------------------------
// Core WMMA 16x16 tile: acc += A[row0:+16, :K] * Bt[col0:+16, :K]^T
// A-frag layout (ISA 7.12.2, 16-bit A 16x32): lane<16 holds M=lane,
//   elems 0..7 -> K=k0..k0+7, 8..15 -> K=k0+16..k0+23; lanes 16..31 get the
//   K=8..15 / 24..31 halves. B-frag: lane holds N col = lane%16, same K
//   order as A (elems 0..7 -> K=k0+8h.., 8..15 -> K=k0+16+8h..).
// ---------------------------------------------------------------------------
__device__ __forceinline__ v8f wmma_tile(const __bf16* __restrict__ A,
                                         const __bf16* __restrict__ Bt,
                                         long lda, long ldb,
                                         int row0, int col0, int K, v8f acc) {
  const int lane = threadIdx.x & 31;
  const int r    = lane & 15;
  const int half = lane >> 4;
  const __bf16* arow = A  + (long)(row0 + r) * lda;
  const __bf16* bcol = Bt + (long)(col0 + r) * ldb;
  for (int k0 = 0; k0 < K; k0 += 32) {
    v16bf a, b;
#pragma unroll
    for (int i = 0; i < 8; ++i) {
      a[i]     = arow[k0 + half * 8 + i];
      a[8 + i] = arow[k0 + 16 + half * 8 + i];
    }
#pragma unroll
    for (int j = 0; j < 8; ++j) {
      b[j]     = bcol[k0 + half * 8 + j];
      b[8 + j] = bcol[k0 + 16 + half * 8 + j];
    }
    acc = __builtin_amdgcn_wmma_f32_16x16x32_bf16(false, a, false, b,
                                                  (short)0, acc, false, false);
  }
  return acc;
}

// Generic GEMM: blockDim=256 (8 waves), wave w -> N-tile blockIdx.x*8+w,
// M-tile blockIdx.y. Guard is wave-uniform (EXEC stays all-ones for WMMA).
__global__ void k_gemm(const __bf16* __restrict__ A, const __bf16* __restrict__ Bt,
                       const float* __restrict__ bias, float* __restrict__ C,
                       int M, int N, int K, int lda) {
  const int wave  = threadIdx.x >> 5;
  const int tileN = blockIdx.x * 8 + wave;
  if (tileN * 16 >= N) return;
  const int row0 = blockIdx.y * 16, col0 = tileN * 16;
  v8f acc = {};
  acc = wmma_tile(A, Bt, lda, K, row0, col0, K, acc);
  const int lane = threadIdx.x & 31;
  const int n  = lane & 15;
  const int mh = (lane >> 4) * 8;
  const float bv = bias ? bias[col0 + n] : 0.f;
#pragma unroll
  for (int v = 0; v < 8; ++v)
    C[(long)(row0 + mh + v) * N + col0 + n] = acc[v] + bv;
}

// Final vocab GEMM: A = h_all (rows = t*64+b), out scattered to [b][t][V],
// masked by active = (t < dec_len[b]); inactive rows forced to 0.
__global__ void k_gemm_fc(const __bf16* __restrict__ A, const __bf16* __restrict__ Bt,
                          const float* __restrict__ bias, const int* __restrict__ dec_len,
                          float* __restrict__ out, int N, int K) {
  const int wave  = threadIdx.x >> 5;
  const int tileN = blockIdx.x * 8 + wave;
  if (tileN * 16 >= N) return;
  const int row0 = blockIdx.y * 16, col0 = tileN * 16;
  v8f acc = {};
  acc = wmma_tile(A, Bt, K, K, row0, col0, K, acc);
  const int lane = threadIdx.x & 31;
  const int n  = lane & 15;
  const int mh = (lane >> 4) * 8;
  const float bv = bias[col0 + n];
#pragma unroll
  for (int v = 0; v < 8; ++v) {
    const int row = row0 + mh + v;
    const int t = row >> 6, b = row & 63;
    const bool act = t < dec_len[b];
    out[((long)b * T_ + t) * (long)V_ + col0 + n] = act ? (acc[v] + bv) : 0.f;
  }
}

// ---------------------------------------------------------------------------
// Prep kernels
// ---------------------------------------------------------------------------

// Stable descending argsort of lengths (B=64, 1 block) + dec_len + out tail.
__global__ void k_sort(const int* __restrict__ cap_len, int* __restrict__ sort_ind,
                       int* __restrict__ dec_len, float* __restrict__ out_tail) {
  __shared__ int lens[B_];
  const int i = threadIdx.x;
  lens[i] = cap_len[i];
  __syncthreads();
  const int li = lens[i];
  int r = 0;
  for (int j = 0; j < B_; ++j) {
    const int lj = lens[j];
    if (lj > li || (lj == li && j < i)) ++r;
  }
  sort_ind[r] = i;
  dec_len[r]  = li - 1;
  out_tail[r]       = (float)(li - 1); // dec_len output
  out_tail[B_ + r]  = (float)i;        // sort_ind output
}

__global__ void k_caps(const int* __restrict__ caps_in, const int* __restrict__ sort_ind,
                       int* __restrict__ caps_s, float* __restrict__ out_caps) {
  const int i = blockIdx.x * 256 + threadIdx.x;
  if (i >= B_ * 52) return;
  const int r = i / 52, j = i % 52;
  const int v = caps_in[sort_ind[r] * 52 + j];
  caps_s[i]   = v;
  out_caps[i] = (float)v;
}

__global__ void k_gather_enc(const float* __restrict__ enc, const int* __restrict__ sort_ind,
                             float* __restrict__ enc_s, __bf16* __restrict__ enc_s_bf) {
  const long i = (long)blockIdx.x * 256 + threadIdx.x;
  if (i >= (long)B_ * P_ * D_) return;
  const int  b   = (int)(i / (P_ * D_));
  const long rem = i % (P_ * D_);
  const float v  = enc[(long)sort_ind[b] * P_ * D_ + rem];
  enc_s[i]    = v;
  enc_s_bf[i] = (__bf16)v;
}

// Mean over P -> first half of init_in (bf16).
__global__ void k_mean(const float* __restrict__ enc_s, __bf16* __restrict__ init_in) {
  const int i = blockIdx.x * 256 + threadIdx.x;
  if (i >= B_ * D_) return;
  const int b = i >> 9, d = i & 511;
  float s = 0.f;
  for (int p = 0; p < P_; ++p) s += enc_s[((long)b * P_ + p) * D_ + d];
  init_in[(long)b * 1024 + d] = (__bf16)(s * (1.f / P_));
}

// Class embedding (NOT sorted, matching reference) -> init_in[512:] and x[1024:1536].
__global__ void k_cls(const float* __restrict__ cls_W, const int* __restrict__ class_k,
                      __bf16* __restrict__ init_in, __bf16* __restrict__ x) {
  const int i = blockIdx.x * 256 + threadIdx.x;
  if (i >= B_ * D_) return;
  const int b = i >> 9, d = i & 511;
  const float v = cls_W[(long)class_k[b] * D_ + d];
  init_in[(long)b * 1024 + 512 + d] = (__bf16)v;
  x[(long)b * 2048 + 1024 + d]      = (__bf16)v;
}

__global__ void k_embs(const float* __restrict__ emb_W, const int* __restrict__ caps_s,
                       __bf16* __restrict__ embs_bf) {
  const long i = (long)blockIdx.x * 256 + threadIdx.x;
  if (i >= (long)B_ * T_ * D_) return;
  const int b = (int)(i / (T_ * D_));
  const int rem = (int)(i % (T_ * D_));
  const int t = rem / D_, d = rem % D_;
  const int tok = caps_s[b * 52 + t];
  embs_bf[i] = (__bf16)emb_W[(long)tok * D_ + d];
}

// Transpose fp32 [K x N] -> bf16 [N x K]   (dst is a Bt operand).
__global__ void k_tr(const float* __restrict__ src, __bf16* __restrict__ dst, int K, int N) {
  const long i = (long)blockIdx.x * 256 + threadIdx.x;
  if (i >= (long)N * K) return;
  const int n = (int)(i / K), k = (int)(i % K);
  dst[i] = (__bf16)src[(long)k * N + n];
}

// Wcat2[n][0:1536] = Wih[n], Wcat2[n][1536:2048] = Whh[n]  (both already row=n, k-contig).
__global__ void k_wcat2(const float* __restrict__ Wih, const float* __restrict__ Whh,
                        __bf16* __restrict__ dst) {
  const long i = (long)blockIdx.x * 256 + threadIdx.x;
  if (i >= (long)2048 * 2048) return;
  const int n = (int)(i >> 11), k = (int)(i & 2047);
  dst[i] = (__bf16)((k < 1536) ? Wih[(long)n * 1536 + k] : Whh[(long)n * 512 + (k - 1536)]);
}

__global__ void k_bias(const float* __restrict__ dec_att_b, const float* __restrict__ f_beta_b,
                       const float* __restrict__ bih, const float* __restrict__ bhh,
                       float* __restrict__ bcat, float* __restrict__ bsum) {
  const int i = blockIdx.x * 256 + threadIdx.x;
  if (i < 2048) bsum[i] = bih[i] + bhh[i];
  if (i < 1024) bcat[i] = (i < 512) ? dec_att_b[i] : f_beta_b[i - 512];
}

// Seed x[:,1536:2048] with bf16(h0).
__global__ void k_postinit(const float* __restrict__ h, __bf16* __restrict__ x) {
  const int i = blockIdx.x * 256 + threadIdx.x;
  if (i >= B_ * D_) return;
  const int b = i >> 9, d = i & 511;
  x[(long)b * 2048 + 1536 + d] = (__bf16)h[i];
}

// ---------------------------------------------------------------------------
// Per-step kernels
// ---------------------------------------------------------------------------

// Attention: e = relu(att1+att2)·w + b0 ; softmax over P ; awe = alpha·enc ;
// awe *= sigmoid(gate_pre) ; fill x[:,0:512]=emb_t, x[:,512:1024]=bf16(awe).
__global__ void k_attention(const float* __restrict__ att1, const float* __restrict__ tmp2,
                            const float* __restrict__ faw, const float* __restrict__ fab,
                            const float* __restrict__ enc_s, const __bf16* __restrict__ embs,
                            __bf16* __restrict__ x, int t) {
  __shared__ float att2_sh[D_];
  __shared__ float e_sh[P_];
  __shared__ float ssum;
  const int b = blockIdx.x, tid = threadIdx.x;
  for (int a = tid; a < D_; a += 256) att2_sh[a] = tmp2[(long)b * 1024 + a];
  __syncthreads();
  for (int p = tid; p < P_; p += 256) {
    const float* row = att1 + ((long)b * P_ + p) * D_;
    float s = 0.f;
    for (int a = 0; a < D_; ++a) {
      const float v = row[a] + att2_sh[a];
      s += (v > 0.f ? v : 0.f) * faw[a];
    }
    e_sh[p] = s + fab[0];
  }
  __syncthreads();
  if (tid == 0) {
    float m = e_sh[0];
    for (int p = 1; p < P_; ++p) m = fmaxf(m, e_sh[p]);
    float s = 0.f;
    for (int p = 0; p < P_; ++p) { const float ev = __expf(e_sh[p] - m); e_sh[p] = ev; s += ev; }
    ssum = s;
  }
  __syncthreads();
  const float inv = 1.f / ssum;
  for (int d = tid; d < D_; d += 256) {
    float acc = 0.f;
    const float* ep = enc_s + (long)b * P_ * D_ + d;
    for (int p = 0; p < P_; ++p) acc += e_sh[p] * ep[(long)p * D_];
    const float gate = sigf(tmp2[(long)b * 1024 + 512 + d]);
    x[(long)b * 2048 + 512 + d] = (__bf16)(gate * acc * inv);
    x[(long)b * 2048 + d]       = embs[((long)b * T_ + t) * D_ + d];
  }
}

// LSTM cell: gates -> (c_new, h_new); masked carry; stash bf16(h_new) for fc
// GEMM and bf16(h_carry) back into x[:,1536:2048] for the next step.
__global__ void k_lstm(const float* __restrict__ gates, const int* __restrict__ dec_len,
                       float* __restrict__ h, float* __restrict__ c,
                       __bf16* __restrict__ x, __bf16* __restrict__ h_all, int t) {
  const int i = blockIdx.x * 256 + threadIdx.x;
  if (i >= B_ * D_) return;
  const int b = i >> 9, d = i & 511;
  const float* g = gates + (long)b * 2048;
  const float ig = sigf(g[d]);
  const float fg = sigf(g[512 + d]);
  const float gg = tanhf(g[1024 + d]);
  const float og = sigf(g[1536 + d]);
  const float cn = fg * c[i] + ig * gg;
  const float hn = og * tanhf(cn);
  h_all[((long)t * B_ + b) * D_ + d] = (__bf16)hn;
  const bool act = t < dec_len[b];
  const float hc = act ? hn : h[i];
  const float cc = act ? cn : c[i];
  h[i] = hc; c[i] = cc;
  x[(long)b * 2048 + 1536 + d] = (__bf16)hc;
}

// ---------------------------------------------------------------------------
// Host driver
// ---------------------------------------------------------------------------
static inline dim3 g1(long n) { return dim3((unsigned)((n + 255) / 256)); }

extern "C" void kernel_launch(void* const* d_in, const int* in_sizes, int n_in,
                              void* d_out, int out_size, void* d_ws, size_t ws_size,
                              hipStream_t stream) {
  (void)in_sizes; (void)n_in; (void)out_size; (void)ws_size;
  const float* encoder_out = (const float*)d_in[0];
  const int*   enc_caps    = (const int*)  d_in[1];
  const int*   cap_len     = (const int*)  d_in[2];
  const int*   class_k     = (const int*)  d_in[3];
  const float* emb_W       = (const float*)d_in[4];
  const float* cls_emb_W   = (const float*)d_in[5];
  const float* enc_att_W   = (const float*)d_in[6];
  const float* enc_att_b   = (const float*)d_in[7];
  const float* dec_att_W   = (const float*)d_in[8];
  const float* dec_att_b   = (const float*)d_in[9];
  const float* full_att_w  = (const float*)d_in[10];
  const float* full_att_b  = (const float*)d_in[11];
  const float* init_h_W    = (const float*)d_in[12];
  const float* init_h_b    = (const float*)d_in[13];
  const float* init_c_W    = (const float*)d_in[14];
  const float* init_c_b    = (const float*)d_in[15];
  const float* f_beta_W    = (const float*)d_in[16];
  const float* f_beta_b    = (const float*)d_in[17];
  const float* lstm_Wih    = (const float*)d_in[18];
  const float* lstm_Whh    = (const float*)d_in[19];
  const float* lstm_bih    = (const float*)d_in[20];
  const float* lstm_bhh    = (const float*)d_in[21];
  const float* fc_W        = (const float*)d_in[22];
  const float* fc_b        = (const float*)d_in[23];

  float* out = (float*)d_out;
  const long PRED_N = (long)B_ * T_ * V_;     // 32,640,000
  float* out_caps = out + PRED_N;             // B*52
  float* out_tail = out_caps + (long)B_ * 52; // dec_len(64) + sort_ind(64)

  // Bump allocator over workspace (256B aligned).
  char* wp = (char*)d_ws;
  auto alloc = [&](size_t bytes) -> void* {
    void* r = (void*)wp;
    wp += (bytes + 255) & ~(size_t)255;
    return r;
  };
  int*    sort_ind = (int*)   alloc(B_ * sizeof(int));
  int*    dec_len  = (int*)   alloc(B_ * sizeof(int));
  int*    caps_s   = (int*)   alloc(B_ * 52 * sizeof(int));
  float*  enc_s    = (float*) alloc((size_t)B_ * P_ * D_ * sizeof(float));
  float*  att1     = (float*) alloc((size_t)B_ * P_ * D_ * sizeof(float));
  float*  tmp2     = (float*) alloc((size_t)B_ * 1024 * sizeof(float));
  float*  gates    = (float*) alloc((size_t)B_ * 2048 * sizeof(float));
  float*  hbuf     = (float*) alloc((size_t)B_ * D_ * sizeof(float));
  float*  cbuf     = (float*) alloc((size_t)B_ * D_ * sizeof(float));
  float*  bcat     = (float*) alloc(1024 * sizeof(float));
  float*  bsum     = (float*) alloc(2048 * sizeof(float));
  __bf16* enc_s_bf = (__bf16*)alloc((size_t)B_ * P_ * D_ * 2);
  __bf16* init_in  = (__bf16*)alloc((size_t)B_ * 1024 * 2);
  __bf16* inithWt  = (__bf16*)alloc((size_t)512 * 1024 * 2);
  __bf16* initcWt  = (__bf16*)alloc((size_t)512 * 1024 * 2);
  __bf16* encattWt = (__bf16*)alloc((size_t)512 * 512 * 2);
  __bf16* Wcat     = (__bf16*)alloc((size_t)1024 * 512 * 2);  // [dec_att^T ; f_beta^T]
  __bf16* Wcat2    = (__bf16*)alloc((size_t)2048 * 2048 * 2); // [Wih | Whh]
  __bf16* fcWt     = (__bf16*)alloc((size_t)V_ * 512 * 2);
  __bf16* embs_bf  = (__bf16*)alloc((size_t)B_ * T_ * D_ * 2);
  __bf16* x_bf     = (__bf16*)alloc((size_t)B_ * 2048 * 2);   // [emb|awe|cls|h]
  __bf16* h_all    = (__bf16*)alloc((size_t)T_ * B_ * D_ * 2);

  // ---- prep ----
  k_sort<<<1, 64, 0, stream>>>(cap_len, sort_ind, dec_len, out_tail);
  k_caps<<<g1(B_ * 52), 256, 0, stream>>>(enc_caps, sort_ind, caps_s, out_caps);
  k_gather_enc<<<g1((long)B_ * P_ * D_), 256, 0, stream>>>(encoder_out, sort_ind, enc_s, enc_s_bf);
  k_mean<<<g1(B_ * D_), 256, 0, stream>>>(enc_s, init_in);
  k_cls<<<g1(B_ * D_), 256, 0, stream>>>(cls_emb_W, class_k, init_in, x_bf);
  k_embs<<<g1((long)B_ * T_ * D_), 256, 0, stream>>>(emb_W, caps_s, embs_bf);
  k_tr<<<g1((long)512 * 1024), 256, 0, stream>>>(init_h_W, inithWt, 1024, 512);
  k_tr<<<g1((long)512 * 1024), 256, 0, stream>>>(init_c_W, initcWt, 1024, 512);
  k_tr<<<g1((long)512 * 512), 256, 0, stream>>>(enc_att_W, encattWt, 512, 512);
  k_tr<<<g1((long)512 * 512), 256, 0, stream>>>(dec_att_W, Wcat, 512, 512);
  k_tr<<<g1((long)512 * 512), 256, 0, stream>>>(f_beta_W, Wcat + (size_t)512 * 512, 512, 512);
  k_tr<<<g1((long)V_ * 512), 256, 0, stream>>>(fc_W, fcWt, 512, V_);
  k_wcat2<<<g1((long)2048 * 2048), 256, 0, stream>>>(lstm_Wih, lstm_Whh, Wcat2);
  k_bias<<<g1(2048), 256, 0, stream>>>(dec_att_b, f_beta_b, lstm_bih, lstm_bhh, bcat, bsum);

  // h0 = init_in @ init_h_W + b ; c0 likewise.  (M=64, N=512, K=1024)
  k_gemm<<<dim3(4, 4), 256, 0, stream>>>(init_in, inithWt, init_h_b, hbuf, 64, 512, 1024, 1024);
  k_gemm<<<dim3(4, 4), 256, 0, stream>>>(init_in, initcWt, init_c_b, cbuf, 64, 512, 1024, 1024);
  // att1 = enc_s @ enc_att_W + b.  (M=12544, N=512, K=512)
  k_gemm<<<dim3(4, 784), 256, 0, stream>>>(enc_s_bf, encattWt, enc_att_b, att1, 12544, 512, 512, 512);
  k_postinit<<<g1(B_ * D_), 256, 0, stream>>>(hbuf, x_bf);

  // ---- 51 recurrent steps ----
  for (int t = 0; t < T_; ++t) {
    // [att2 | gate_pre] = h @ [dec_att_W | f_beta_W] + [b|b]  (A = x cols 1536.., lda=2048)
    k_gemm<<<dim3(8, 4), 256, 0, stream>>>(x_bf + 1536, Wcat, bcat, tmp2, 64, 1024, 512, 2048);
    k_attention<<<B_, 256, 0, stream>>>(att1, tmp2, full_att_w, full_att_b, enc_s, embs_bf, x_bf, t);
    // gates = [emb,awe,cls,h] @ [Wih|Whh]^T + (bih+bhh)   (M=64, N=2048, K=2048)
    k_gemm<<<dim3(16, 4), 256, 0, stream>>>(x_bf, Wcat2, bsum, gates, 64, 2048, 2048, 2048);
    k_lstm<<<g1(B_ * D_), 256, 0, stream>>>(gates, dec_len, hbuf, cbuf, x_bf, h_all, t);
  }

  // ---- batched vocab projection, masked + scattered to (B, T, V) ----
  // M = T*B = 3264 rows (t-major), N = 10000, K = 512.
  k_gemm_fc<<<dim3(79, 204), 256, 0, stream>>>(h_all, fcWt, fc_b, dec_len, out, V_, 512);
}
